// Seq2SeqPointAttnCE_1288490189165
// MI455X (gfx1250) — compile-verified, matrix operand order checked by reading, not run on hardware
//
#include <hip/hip_runtime.h>
#include <hip/hip_bf16.h>

typedef _Float16 v8h  __attribute__((ext_vector_type(8)));
typedef _Float16 v16h __attribute__((ext_vector_type(16)));
typedef float    v8f  __attribute__((ext_vector_type(8)));

// Sizes (compile-time constants for this problem)
#define BB 8
#define SS 128
#define TT 24
#define EE 512
#define HH 512
#define LL 256     // 2*S
#define G3 1536    // 3*H
#define DD 2048    // 3*H + E

// ---------------------------------------------------------------------------
// WMMA helpers: V_WMMA_F32_16X16X32_F16, wave32.
// A fragment: lane lo (=lane&15) holds row M=lo; element e in [0,16):
//   K = ((e>>3)<<4) + (lane>>4)*8 + (e&7)  -> two contiguous 8-half chunks.
// B fragment: weight stored row-major (N x K); lane lo holds row n=lo with the
//   same K-chunk pattern.
// C/D: M = r + (lane>>4)*8, N = lane&15.
// ---------------------------------------------------------------------------
__device__ __forceinline__ v16h frag_load(const _Float16* __restrict__ row_k0, int hi) {
  const _Float16* p = row_k0 + hi * 8;
  v8h a = *(const v8h*)(p);
  v8h b = *(const v8h*)(p + 16);
  v16h r;
#pragma unroll
  for (int i = 0; i < 8; ++i) { r[i] = a[i]; r[i + 8] = b[i]; }
  return r;
}

__device__ __forceinline__ v8f wmma_f32_f16(v16h a, v16h b, v8f c) {
  return __builtin_amdgcn_wmma_f32_16x16x32_f16(false, a, false, b, (short)0, c,
                                                false, false);
}

// ---------------------------------------------------------------------------
// Wave-tiled GEMM with N-register-blocking: each wave computes a 16x64 strip
// (4 n-tiles), reusing the A fragment across 4 WMMAs per k-step.
// C(MxN) = A16(MxK) @ W16(NxK)^T + bias[n].  N multiple of 64, K mult of 32.
// ---------------------------------------------------------------------------
__global__ void gemm_f16w(const _Float16* __restrict__ A, int lda,
                          const _Float16* __restrict__ W, int ldw,
                          const float* __restrict__ bias,
                          float* __restrict__ C, int ldc,
                          int Mtiles, int Nstrips, int K) {
  int wid = (blockIdx.x * blockDim.x + threadIdx.x) >> 5;
  if (wid >= Mtiles * Nstrips) return;  // wave-uniform exit; EXEC stays full
  int lane = threadIdx.x & 31;
  int lo = lane & 15, hi = lane >> 4;
  int tm = wid / Nstrips, tn0 = (wid % Nstrips) * 4;
  const _Float16* arow = A + (size_t)(tm * 16 + lo) * lda;
  const _Float16* wrow = W + (size_t)(tn0 * 16 + lo) * ldw;
  v8f acc[4] = {};
  for (int k0 = 0; k0 < K; k0 += 32) {
    v16h af = frag_load(arow + k0, hi);
#pragma unroll
    for (int j = 0; j < 4; ++j) {
      v16h bf = frag_load(wrow + (size_t)j * 16 * ldw + k0, hi);
      acc[j] = wmma_f32_f16(af, bf, acc[j]);
    }
  }
#pragma unroll
  for (int j = 0; j < 4; ++j) {
    float bv = bias[(tn0 + j) * 16 + lo];
#pragma unroll
    for (int r = 0; r < 8; ++r) {
      C[(size_t)(tm * 16 + r + hi * 8) * ldc + (tn0 + j) * 16 + lo] = acc[j][r] + bv;
    }
  }
}

// ---------------------------------------------------------------------------
// Small prep kernels
// ---------------------------------------------------------------------------
__global__ void f32_to_f16(const float* __restrict__ s, _Float16* __restrict__ d, int n) {
  for (int i = blockIdx.x * blockDim.x + threadIdx.x; i < n; i += gridDim.x * blockDim.x)
    d[i] = (_Float16)s[i];
}

__global__ void split_attn(const float* __restrict__ attn_W,
                           _Float16* __restrict__ Whid, _Float16* __restrict__ Wenc) {
  int i = blockIdx.x * blockDim.x + threadIdx.x;
  if (i >= HH * HH) return;
  int d = i >> 9, h = i & 511;
  Whid[i] = (_Float16)attn_W[d * (2 * HH) + h];
  Wenc[i] = (_Float16)attn_W[d * (2 * HH) + HH + h];
}

__global__ void gather_emb(const int* __restrict__ seq, const float* __restrict__ emb,
                           _Float16* __restrict__ X, int rows) {
  int i = blockIdx.x * blockDim.x + threadIdx.x;
  if (i >= rows * EE) return;
  int m = i >> 9, k = i & 511;
  X[i] = (_Float16)emb[(size_t)seq[m] * EE + k];
}

// w_eff[k] = sum_d fc_out_W[d] * fc_hid_W[d,k];  c0 = fc_out_b + fc_out_W.fc_hid_b
__global__ void weff_kernel(const float* __restrict__ fc_hid_W,
                            const float* __restrict__ fc_out_W,
                            const float* __restrict__ fc_hid_b,
                            const float* __restrict__ fc_out_b,
                            float* __restrict__ w_eff, float* __restrict__ c0) {
  int k = blockIdx.x * blockDim.x + threadIdx.x;
  if (k < DD) {
    float acc = 0.f;
    for (int d = 0; d < DD; ++d) acc += fc_out_W[d] * fc_hid_W[(size_t)d * DD + k];
    w_eff[k] = acc;
  }
  if (k == 0) {
    float acc = fc_out_b[0];
    for (int d = 0; d < DD; ++d) acc += fc_out_W[d] * fc_hid_b[d];
    *c0 = acc;
  }
}

// base[b*L+l] = c0 + enc_out[b,l,:] . w_eff[0:H]
__global__ void base_kernel(const float* __restrict__ enc_out,
                            const float* __restrict__ w_eff,
                            const float* __restrict__ c0, float* __restrict__ base) {
  int i = blockIdx.x * blockDim.x + threadIdx.x;
  if (i >= BB * LL) return;
  const float* row = enc_out + (size_t)i * HH;
  float acc = *c0;
  for (int j = 0; j < HH; ++j) acc += row[j] * w_eff[j];
  base[i] = acc;
}

// h0 = tanh(concat(pre_h,post_h) @ fc_enc_W^T + fc_enc_b)
__global__ void fc_enc_kernel(const float* __restrict__ enc_out,
                              const float* __restrict__ W, const float* __restrict__ bias,
                              float* __restrict__ h0) {
  int i = blockIdx.x * blockDim.x + threadIdx.x;
  if (i >= BB * HH) return;
  int b = i >> 9, n = i & 511;
  const float* wr = W + (size_t)n * (2 * HH);
  const float* pre = enc_out + (size_t)(b * LL + (SS - 1)) * HH;
  const float* post = enc_out + (size_t)(b * LL + (LL - 1)) * HH;
  float acc = bias[n];
  for (int k = 0; k < HH; ++k) acc += pre[k] * wr[k] + post[k] * wr[HH + k];
  h0[i] = tanhf(acc);
}

// ---------------------------------------------------------------------------
// Encoder GRU scan: one 1024-thread workgroup (32 waves), h kept in LDS,
// gh = h @ Whh^T done with WMMA each step (gi precomputed by gemm_f16w).
// Each wave owns 3 n-tiles; A fragment loaded once per k-step and reused 3x.
// ---------------------------------------------------------------------------
__global__ void __launch_bounds__(1024) encoder_scan(
    const float* __restrict__ gi,        // (B*S) x 3H, bias bih already added
    const _Float16* __restrict__ Whh,    // 3H x H (row-major N x K)
    const float* __restrict__ bhh,       // 3H
    float* __restrict__ enc_out,         // (B*L) x H
    _Float16* __restrict__ enc_out16,    // (B*L) x H
    int loff) {
  __shared__ _Float16 hf16[16 * HH];     // A-matrix staging, rows 8..15 = 0
  __shared__ float ghL[BB * G3];
  int tid = threadIdx.x;
  int wave = tid >> 5, lane = tid & 31, lo = lane & 15, hi = lane >> 4;

  for (int i = tid; i < 16 * HH; i += 1024) hf16[i] = (_Float16)0.f;
  float hreg[4] = {0.f, 0.f, 0.f, 0.f};
  __syncthreads();

  int tn = wave * 3;
  const _Float16* wrow = Whh + (size_t)(tn * 16 + lo) * HH;
  const _Float16* arow = hf16 + lo * HH;

  for (int s = 0; s < SS; ++s) {
    // gh = h @ Whh^T : 96 n-tiles over 32 waves, 3 per wave with A reuse
    v8f acc[3] = {};
    for (int k0 = 0; k0 < HH; k0 += 32) {
      v16h af = frag_load(arow + k0, hi);
#pragma unroll
      for (int tt = 0; tt < 3; ++tt) {
        v16h bf = frag_load(wrow + (size_t)tt * 16 * HH + k0, hi);
        acc[tt] = wmma_f32_f16(af, bf, acc[tt]);
      }
    }
    if (hi == 0) {
#pragma unroll
      for (int tt = 0; tt < 3; ++tt) {
        float bv = bhh[(tn + tt) * 16 + lo];
#pragma unroll
        for (int r = 0; r < 8; ++r) ghL[r * G3 + (tn + tt) * 16 + lo] = acc[tt][r] + bv;
      }
    }
    __syncthreads();
    // gate math + h update
#pragma unroll
    for (int p = 0; p < 4; ++p) {
      int e = tid + p * 1024;
      int b = e >> 9, j = e & 511;
      const float* gr = gi + (size_t)(b * SS + s) * G3;
      float ir = gr[j], iz = gr[HH + j], inn = gr[2 * HH + j];
      float hr = ghL[b * G3 + j], hz = ghL[b * G3 + HH + j], hn = ghL[b * G3 + 2 * HH + j];
      float r = 1.f / (1.f + __expf(-(ir + hr)));
      float z = 1.f / (1.f + __expf(-(iz + hz)));
      float n = tanhf(inn + r * hn);
      float h = (1.f - z) * n + z * hreg[p];
      hreg[p] = h;
      hf16[b * HH + j] = (_Float16)h;
      size_t oidx = (size_t)(b * LL + loff + s) * HH + j;
      enc_out[oidx] = h;
      enc_out16[oidx] = (_Float16)h;
    }
    __syncthreads();
  }
}

// ---------------------------------------------------------------------------
// Decoder: single 1024-thread workgroup, 24 serial steps.
// Per step: q=h@Whid^T (WMMA) -> energy/softmax -> weighted -> GRU cell
// (gi = [e,weighted]@Wih_dec^T, gh = h@Whh_dec^T, both WMMA, N-blocked with
// A-fragment reuse) -> collapsed output logits via w_eff.
// ---------------------------------------------------------------------------
__global__ void __launch_bounds__(1024) decoder_kernel(
    const int* __restrict__ pre_seq, const int* __restrict__ trg,
    const float* __restrict__ emb,
    const _Float16* __restrict__ Whid16, const float* __restrict__ attn_v,
    const float* __restrict__ enc_proj, const float* __restrict__ enc_out,
    const _Float16* __restrict__ Wih_dec16, const _Float16* __restrict__ Whh_dec16,
    const float* __restrict__ bih_dec, const float* __restrict__ bhh_dec,
    const float* __restrict__ w_eff, const float* __restrict__ base,
    const float* __restrict__ h0, float* __restrict__ out) {
  __shared__ _Float16 hf16[16 * HH];     // 16 KB
  __shared__ _Float16 x16[16 * 1024];    // 32 KB, rows 8..15 = 0
  __shared__ float h32[BB * HH];         // 16 KB
  __shared__ float e32[BB * EE];         // 16 KB
  __shared__ float qv[BB * HH];          // 16 KB
  __shared__ float wsum[BB * HH];        // 16 KB
  __shared__ float sc[BB * LL];          // 8 KB
  __shared__ float giL[BB * G3];         // 48 KB
  __shared__ float ghL[BB * G3];         // 48 KB
  __shared__ float vbuf[HH];             // 2 KB (attn_v cache)
  __shared__ float red[1024];
  __shared__ float Sb[BB];

  int tid = threadIdx.x;
  int wave = tid >> 5, lane = tid & 31, lo = lane & 15, hi = lane >> 4;

  for (int i = tid; i < 16 * HH; i += 1024) hf16[i] = (_Float16)0.f;
  for (int i = tid; i < 16 * 1024; i += 1024) x16[i] = (_Float16)0.f;
  if (tid < HH) vbuf[tid] = attn_v[tid];
  __syncthreads();
  for (int i = tid; i < BB * HH; i += 1024) {
    float h = h0[i];
    h32[i] = h;
    hf16[i] = (_Float16)h;   // rows 0..7
  }
  __syncthreads();

  for (int t = 0; t < TT; ++t) {
    // --- A: token embedding
    for (int i = tid; i < BB * EE; i += 1024) {
      int b = i >> 9, k = i & 511;
      int tok = (t == 0) ? pre_seq[b * SS + (SS - 1)] : trg[b * TT + t - 1];
      e32[i] = emb[(size_t)tok * EE + k];
    }
    // --- B: q = h @ Whid^T  (32 n-tiles, one per wave)
    {
      int tn = wave;
      const _Float16* wrow = Whid16 + (size_t)(tn * 16 + lo) * HH;
      const _Float16* arow = hf16 + lo * HH;
      v8f acc = {};
      for (int k0 = 0; k0 < HH; k0 += 32)
        acc = wmma_f32_f16(frag_load(arow + k0, hi), frag_load(wrow + k0, hi), acc);
      if (hi == 0) {
#pragma unroll
        for (int r = 0; r < 8; ++r) qv[r * HH + tn * 16 + lo] = acc[r];
      }
    }
    __syncthreads();
    // --- C: attention scores
    for (int i = tid; i < BB * LL; i += 1024) {
      int b = i >> 8, l = i & 255;
      const float* ep = enc_proj + (size_t)(b * LL + l) * HH;
      const float* qb = qv + b * HH;
      float s = 0.f;
      for (int j = 0; j < HH; ++j) s += tanhf(ep[j] + qb[j]) * vbuf[j];
      sc[i] = s;
    }
    __syncthreads();
    // --- D: softmax over L per batch row
    if (tid < BB) {
      float m = -1e30f;
      for (int l = 0; l < LL; ++l) m = fmaxf(m, sc[tid * LL + l]);
      float sum = 0.f;
      for (int l = 0; l < LL; ++l) {
        float ev = __expf(sc[tid * LL + l] - m);
        sc[tid * LL + l] = ev;
        sum += ev;
      }
      float inv = 1.f / sum;
      for (int l = 0; l < LL; ++l) sc[tid * LL + l] *= inv;
    }
    __syncthreads();
    // --- E: weighted = a @ enc_out
    for (int i = tid; i < BB * HH; i += 1024) {
      int b = i >> 9, j = i & 511;
      const float* eo = enc_out + (size_t)b * LL * HH + j;
      const float* ab = sc + b * LL;
      float acc = 0.f;
      for (int l = 0; l < LL; ++l) acc += ab[l] * eo[(size_t)l * HH];
      wsum[i] = acc;
    }
    __syncthreads();
    // --- F: build x = concat(e, weighted) as f16 A-matrix
    for (int i = tid; i < BB * HH; i += 1024) {
      int b = i >> 9, j = i & 511;
      x16[b * 1024 + j] = (_Float16)e32[i];
      x16[b * 1024 + HH + j] = (_Float16)wsum[i];
    }
    __syncthreads();
    // --- G: gi = x @ Wih_dec^T (+bih), gh = h @ Whh_dec^T (+bhh)
    {
      int tn = wave * 3;
      // gi: A = x16 (lda 1024), K = 1024
      {
        const _Float16* wrow = Wih_dec16 + (size_t)(tn * 16 + lo) * 1024;
        const _Float16* arow = x16 + lo * 1024;
        v8f acc[3] = {};
        for (int k0 = 0; k0 < 1024; k0 += 32) {
          v16h af = frag_load(arow + k0, hi);
#pragma unroll
          for (int tt = 0; tt < 3; ++tt)
            acc[tt] = wmma_f32_f16(af, frag_load(wrow + (size_t)tt * 16 * 1024 + k0, hi),
                                   acc[tt]);
        }
        if (hi == 0) {
#pragma unroll
          for (int tt = 0; tt < 3; ++tt) {
            float bv = bih_dec[(tn + tt) * 16 + lo];
#pragma unroll
            for (int r = 0; r < 8; ++r)
              giL[r * G3 + (tn + tt) * 16 + lo] = acc[tt][r] + bv;
          }
        }
      }
      // gh: A = hf16 (lda 512), K = 512
      {
        const _Float16* wrow = Whh_dec16 + (size_t)(tn * 16 + lo) * HH;
        const _Float16* arow = hf16 + lo * HH;
        v8f acc[3] = {};
        for (int k0 = 0; k0 < HH; k0 += 32) {
          v16h af = frag_load(arow + k0, hi);
#pragma unroll
          for (int tt = 0; tt < 3; ++tt)
            acc[tt] = wmma_f32_f16(af, frag_load(wrow + (size_t)tt * 16 * HH + k0, hi),
                                   acc[tt]);
        }
        if (hi == 0) {
#pragma unroll
          for (int tt = 0; tt < 3; ++tt) {
            float bv = bhh_dec[(tn + tt) * 16 + lo];
#pragma unroll
            for (int r = 0; r < 8; ++r)
              ghL[r * G3 + (tn + tt) * 16 + lo] = acc[tt][r] + bv;
          }
        }
      }
    }
    __syncthreads();
    // --- H: GRU gates, h update
    for (int i = tid; i < BB * HH; i += 1024) {
      int b = i >> 9, j = i & 511;
      float ir = giL[b * G3 + j], iz = giL[b * G3 + HH + j], inn = giL[b * G3 + 2 * HH + j];
      float hr = ghL[b * G3 + j], hz = ghL[b * G3 + HH + j], hn = ghL[b * G3 + 2 * HH + j];
      float r = 1.f / (1.f + __expf(-(ir + hr)));
      float z = 1.f / (1.f + __expf(-(iz + hz)));
      float n = tanhf(inn + r * hn);
      float h = (1.f - z) * n + z * h32[i];
      h32[i] = h;
      hf16[i] = (_Float16)h;
    }
    __syncthreads();
    // --- I: per-batch scalar S[b] = weighted.w2 + h2.w3 + e.w4
    {
      int b = tid >> 7, j0 = tid & 127;
      float acc = 0.f;
      for (int j = j0; j < HH; j += 128) {
        acc += wsum[b * HH + j] * w_eff[HH + j] + h32[b * HH + j] * w_eff[2 * HH + j] +
               e32[b * HH + j] * w_eff[3 * HH + j];
      }
      red[tid] = acc;
    }
    __syncthreads();
    for (int off = 64; off >= 1; off >>= 1) {
      if ((tid & 127) < off) red[tid] += red[tid + off];
      __syncthreads();
    }
    if ((tid & 127) == 0) Sb[tid >> 7] = red[tid];
    __syncthreads();
    // --- J: logits[b,t,l] = base[b,l] + S[b]; split (pre|post) outputs
    for (int i = tid; i < BB * LL; i += 1024) {
      int b = i >> 8, l = i & 255;
      float val = base[b * LL + l] + Sb[b];
      int idx = (l < SS) ? ((b * TT + t) * SS + l)
                         : (BB * TT * SS + (b * TT + t) * SS + (l - SS));
      out[idx] = val;
    }
    __syncthreads();
  }
}

// ---------------------------------------------------------------------------
extern "C" void kernel_launch(void* const* d_in, const int* in_sizes, int n_in,
                              void* d_out, int out_size, void* d_ws, size_t ws_size,
                              hipStream_t stream) {
  (void)in_sizes; (void)n_in; (void)out_size; (void)ws_size;
  const int*   pre_seq  = (const int*)d_in[0];
  const int*   post_seq = (const int*)d_in[1];
  const int*   trg      = (const int*)d_in[2];
  const float* emb      = (const float*)d_in[3];
  const float* Wih_pre  = (const float*)d_in[4];
  const float* Whh_pre  = (const float*)d_in[5];
  const float* bih_pre  = (const float*)d_in[6];
  const float* bhh_pre  = (const float*)d_in[7];
  const float* Wih_post = (const float*)d_in[8];
  const float* Whh_post = (const float*)d_in[9];
  const float* bih_post = (const float*)d_in[10];
  const float* bhh_post = (const float*)d_in[11];
  const float* fc_enc_W = (const float*)d_in[12];
  const float* fc_enc_b = (const float*)d_in[13];
  const float* attn_W   = (const float*)d_in[14];
  const float* attn_b   = (const float*)d_in[15];
  const float* attn_v   = (const float*)d_in[16];
  const float* Wih_dec  = (const float*)d_in[17];
  const float* Whh_dec  = (const float*)d_in[18];
  const float* bih_dec  = (const float*)d_in[19];
  const float* bhh_dec  = (const float*)d_in[20];
  const float* fc_hid_W = (const float*)d_in[21];
  const float* fc_hid_b = (const float*)d_in[22];
  const float* fc_out_W = (const float*)d_in[23];
  const float* fc_out_b = (const float*)d_in[24];
  float* out = (float*)d_out;

  char* ws = (char*)d_ws;
  size_t off = 0;
  auto alloc = [&](size_t bytes) -> void* {
    void* p = ws + off;
    off += (bytes + 255) & ~(size_t)255;
    return p;
  };

  _Float16* Wih_pre16  = (_Float16*)alloc((size_t)G3 * EE * 2);
  _Float16* Whh_pre16  = (_Float16*)alloc((size_t)G3 * HH * 2);
  _Float16* Wih_post16 = (_Float16*)alloc((size_t)G3 * EE * 2);
  _Float16* Whh_post16 = (_Float16*)alloc((size_t)G3 * HH * 2);
  _Float16* Wih_dec16  = (_Float16*)alloc((size_t)G3 * 1024 * 2);
  _Float16* Whh_dec16  = (_Float16*)alloc((size_t)G3 * HH * 2);
  _Float16* Whid16     = (_Float16*)alloc((size_t)HH * HH * 2);
  _Float16* Wenc16     = (_Float16*)alloc((size_t)HH * HH * 2);
  _Float16* Xpre16     = (_Float16*)alloc((size_t)BB * SS * EE * 2);
  _Float16* Xpost16    = (_Float16*)alloc((size_t)BB * SS * EE * 2);
  float* gi_pre   = (float*)alloc((size_t)BB * SS * G3 * 4);
  float* gi_post  = (float*)alloc((size_t)BB * SS * G3 * 4);
  float* enc_out  = (float*)alloc((size_t)BB * LL * HH * 4);
  _Float16* enc_out16 = (_Float16*)alloc((size_t)BB * LL * HH * 2);
  float* enc_proj = (float*)alloc((size_t)BB * LL * HH * 4);
  float* w_eff    = (float*)alloc((size_t)DD * 4);
  float* c0       = (float*)alloc(4);
  float* base     = (float*)alloc((size_t)BB * LL * 4);
  float* h0       = (float*)alloc((size_t)BB * HH * 4);

  // --- weight conversions
  f32_to_f16<<<1024, 256, 0, stream>>>(Wih_pre,  Wih_pre16,  G3 * EE);
  f32_to_f16<<<1024, 256, 0, stream>>>(Whh_pre,  Whh_pre16,  G3 * HH);
  f32_to_f16<<<1024, 256, 0, stream>>>(Wih_post, Wih_post16, G3 * EE);
  f32_to_f16<<<1024, 256, 0, stream>>>(Whh_post, Whh_post16, G3 * HH);
  f32_to_f16<<<2048, 256, 0, stream>>>(Wih_dec,  Wih_dec16,  G3 * 1024);
  f32_to_f16<<<1024, 256, 0, stream>>>(Whh_dec,  Whh_dec16,  G3 * HH);
  split_attn<<<(HH * HH + 255) / 256, 256, 0, stream>>>(attn_W, Whid16, Wenc16);
  gather_emb<<<(BB * SS * EE + 255) / 256, 256, 0, stream>>>(pre_seq, emb, Xpre16, BB * SS);
  gather_emb<<<(BB * SS * EE + 255) / 256, 256, 0, stream>>>(post_seq, emb, Xpost16, BB * SS);
  weff_kernel<<<(DD + 255) / 256, 256, 0, stream>>>(fc_hid_W, fc_out_W, fc_hid_b, fc_out_b,
                                                    w_eff, c0);

  // --- hoisted input-side GRU GEMMs: gi = X @ Wih^T + bih
  {
    int strips = (BB * SS / 16) * (G3 / 64);              // 64 * 24
    int blocks = (strips * 32 + 255) / 256;
    gemm_f16w<<<blocks, 256, 0, stream>>>(Xpre16, EE, Wih_pre16, EE, bih_pre,
                                          gi_pre, G3, BB * SS / 16, G3 / 64, EE);
    gemm_f16w<<<blocks, 256, 0, stream>>>(Xpost16, EE, Wih_post16, EE, bih_post,
                                          gi_post, G3, BB * SS / 16, G3 / 64, EE);
  }

  // --- serial encoder scans (persistent single-WG, WMMA for h@Whh^T)
  encoder_scan<<<1, 1024, 0, stream>>>(gi_pre, Whh_pre16, bhh_pre, enc_out, enc_out16, 0);
  encoder_scan<<<1, 1024, 0, stream>>>(gi_post, Whh_post16, bhh_post, enc_out, enc_out16, SS);

  // --- enc_proj = enc_out @ W_enc^T + attn_b
  {
    int strips = (BB * LL / 16) * (HH / 64);              // 128 * 8
    int blocks = (strips * 32 + 255) / 256;
    gemm_f16w<<<blocks, 256, 0, stream>>>(enc_out16, HH, Wenc16, HH, attn_b,
                                          enc_proj, HH, BB * LL / 16, HH / 64, HH);
  }

  // --- decoder init hidden + collapsed-output base
  fc_enc_kernel<<<(BB * HH + 255) / 256, 256, 0, stream>>>(enc_out, fc_enc_W, fc_enc_b, h0);
  base_kernel<<<(BB * LL + 255) / 256, 256, 0, stream>>>(enc_out, w_eff, c0, base);

  // --- serial decoder (persistent single-WG)
  decoder_kernel<<<1, 1024, 0, stream>>>(pre_seq, trg, emb, Whid16, attn_v, enc_proj,
                                         enc_out, Wih_dec16, Whh_dec16, bih_dec, bhh_dec,
                                         w_eff, base, h0, out);
}
